// GINENet_82282983457236
// MI455X (gfx1250) — compile-verified, matrix operand order checked by reading, not run on hardware
//
#include <hip/hip_runtime.h>
#include <math.h>

// ---------------------------------------------------------------------------
// CDNA5 (gfx1250) fused GINE network.
// wave32; WMMA bf16 16x16x32 with f32 accumulate for every GEMM.
// ---------------------------------------------------------------------------

typedef __attribute__((ext_vector_type(16))) __bf16 v16bf;
typedef __attribute__((ext_vector_type(8)))  float  v8f;

#define WAVES   8
#define TILE_M  16
#define BLOCK_M (WAVES * TILE_M)   // 128 rows (edges/nodes) per 256-thread block
#define H       64

__device__ __forceinline__ unsigned short f2bf(float f) {
  union { float f; unsigned u; } c; c.f = f;
  unsigned u = c.u;
  u += 0x7FFFu + ((u >> 16) & 1u);     // round-to-nearest-even
  return (unsigned short)(u >> 16);
}

union FragBF { v16bf v; unsigned short u[16]; };

// ISA 7.12.2 16-bit A/B fragment: element e of lane(half) maps to this K
// within a 32-wide K block.
__device__ __forceinline__ int kmap(int e, int hf) {
  return (e & 7) | ((e & 8) << 1) | (hf << 3);
}

__device__ __forceinline__ v8f wmma_bf16(const FragBF& A, const FragBF& B, v8f C) {
  return __builtin_amdgcn_wmma_f32_16x16x32_bf16(
      /*neg_a=*/false, A.v, /*neg_b=*/false, B.v,
      /*c_mod=*/(short)0, C, /*reuse_a=*/false, /*reuse_b=*/false);
}

// ---------------------------------------------------------------------------
// zero fill
// ---------------------------------------------------------------------------
__global__ void zero_f32(float* __restrict__ p, long long n) {
  long long i = (long long)blockIdx.x * blockDim.x + threadIdx.x;
  if (i < n) p[i] = 0.0f;
}

// ---------------------------------------------------------------------------
// Fused: edge MLP (WMMA) -> msg = relu(h[src] + emb) -> atomic scatter-add.
// edge_emb is recomputed per conv; it never hits global memory.
// ---------------------------------------------------------------------------
__global__ void __launch_bounds__(256)
edge_msg_scatter(const float* __restrict__ h,      // [N,64]
                 const int*   __restrict__ src,
                 const int*   __restrict__ dst,
                 const float* __restrict__ attr,   // [E]
                 const float* __restrict__ w1,     // [64]
                 const float* __restrict__ b1,     // [64]
                 const float* __restrict__ w2,     // [64,64] (k,n) row-major
                 const float* __restrict__ b2,     // [64]
                 float* __restrict__ aggr,         // [N,64]
                 int E) {
  __shared__ unsigned short w2t[H * H];             // bf16, [n][k]
  __shared__ unsigned short h1s[WAVES][TILE_M * H]; // bf16, per-wave [16][64]
  __shared__ int sidx[WAVES][TILE_M];
  __shared__ int didx[WAVES][TILE_M];

  const int tid = threadIdx.x;

  // stage w2^T as bf16 (tiny; L2-resident across blocks)
  for (int i = tid; i < H * H; i += blockDim.x) {
    int k = i & 63, n = i >> 6;
    w2t[i] = f2bf(w2[k * H + n]);
  }

  const int wave = tid >> 5;
  const int lane = tid & 31;
  const int hf   = lane >> 4;
  const int l16  = lane & 15;
  const long long e0 = (long long)blockIdx.x * BLOCK_M + wave * TILE_M;

  // stage edge indices for this wave's 16 edges
  {
    long long e = e0 + l16; if (e >= E) e = E - 1;
    if (hf == 0) sidx[wave][l16] = src[e];
    else         didx[wave][l16] = dst[e];
  }

  // h1 = relu(attr * w1 + b1): lane covers row l16, cols hf*32 .. +31
  {
    long long e = e0 + l16; if (e >= E) e = E - 1;
    float a = attr[e];
    int cb = hf * 32;
    #pragma unroll 8
    for (int c = 0; c < 32; ++c) {
      float v = fmaf(a, w1[cb + c], b1[cb + c]);
      h1s[wave][l16 * H + cb + c] = f2bf(v > 0.0f ? v : 0.0f);
    }
  }
  __syncthreads();

  // emb(16x64) = h1(16x64) @ w2(64x64) + b2   -- 8 WMMAs per wave
  v8f acc[4];
  #pragma unroll
  for (int nb = 0; nb < 4; ++nb) {
    float bias = b2[nb * 16 + l16];
    #pragma unroll
    for (int r = 0; r < 8; ++r) acc[nb][r] = bias;
    #pragma unroll
    for (int kb = 0; kb < 2; ++kb) {
      FragBF A, B;
      #pragma unroll
      for (int e2 = 0; e2 < 16; ++e2) {
        int k = kb * 32 + kmap(e2, hf);
        A.u[e2] = h1s[wave][l16 * H + k];
        B.u[e2] = w2t[(nb * 16 + l16) * H + k];
      }
      acc[nb] = wmma_bf16(A, B, acc[nb]);
    }
  }

  // msg = relu(h[src] + emb); scatter-add to aggr[dst].
  // D layout: VGPR r -> row M = r + 8*hf, col N = nb*16 + l16.
  #pragma unroll
  for (int nb = 0; nb < 4; ++nb) {
    int n = nb * 16 + l16;
    #pragma unroll
    for (int r = 0; r < 8; ++r) {
      int m = r + 8 * hf;
      long long e = e0 + m;
      if (e < E) {
        float msg = acc[nb][r] + h[(long long)sidx[wave][m] * H + n];
        msg = msg > 0.0f ? msg : 0.0f;
        atomicAdd(&aggr[(long long)didx[wave][m] * H + n], msg);
      }
    }
  }
}

// ---------------------------------------------------------------------------
// Node update: out = relu( relu(u@w1+b1) @ w2 + b2 ),  u = (1+eps)*h + aggr.
// 16 WMMAs per wave.
// ---------------------------------------------------------------------------
__global__ void __launch_bounds__(256)
node_mlp(const float* __restrict__ hin,   // [N,64]
         const float* __restrict__ aggr,  // [N,64]
         const float* __restrict__ eps_p, // scalar
         const float* __restrict__ w1, const float* __restrict__ b1,
         const float* __restrict__ w2, const float* __restrict__ b2,
         float* __restrict__ hout, int N) {
  __shared__ unsigned short w1t[H * H];              // bf16 [n][k]
  __shared__ unsigned short w2t[H * H];              // bf16 [n][k]
  __shared__ unsigned short tA[WAVES][TILE_M * H];   // u tile
  __shared__ unsigned short tT[WAVES][TILE_M * H];   // hidden tile

  const int tid = threadIdx.x;
  for (int i = tid; i < H * H; i += blockDim.x) {
    int k = i & 63, n = i >> 6;
    w1t[i] = f2bf(w1[k * H + n]);
    w2t[i] = f2bf(w2[k * H + n]);
  }

  const int wave = tid >> 5;
  const int lane = tid & 31;
  const int hf   = lane >> 4;
  const int l16  = lane & 15;
  const long long n0 = (long long)blockIdx.x * BLOCK_M + wave * TILE_M;
  const float epp = 1.0f + eps_p[0];

  // u tile
  {
    long long node = n0 + l16; if (node >= N) node = N - 1;
    int cb = hf * 32;
    #pragma unroll 8
    for (int c = 0; c < 32; ++c) {
      float v = fmaf(epp, hin[node * H + cb + c], aggr[node * H + cb + c]);
      tA[wave][l16 * H + cb + c] = f2bf(v);
    }
  }
  __syncthreads();

  v8f acc[4];
  // layer 1: t = relu(u @ w1 + b1)
  #pragma unroll
  for (int nb = 0; nb < 4; ++nb) {
    float bias = b1[nb * 16 + l16];
    #pragma unroll
    for (int r = 0; r < 8; ++r) acc[nb][r] = bias;
    #pragma unroll
    for (int kb = 0; kb < 2; ++kb) {
      FragBF A, B;
      #pragma unroll
      for (int e2 = 0; e2 < 16; ++e2) {
        int k = kb * 32 + kmap(e2, hf);
        A.u[e2] = tA[wave][l16 * H + k];
        B.u[e2] = w1t[(nb * 16 + l16) * H + k];
      }
      acc[nb] = wmma_bf16(A, B, acc[nb]);
    }
  }
  #pragma unroll
  for (int nb = 0; nb < 4; ++nb)
    #pragma unroll
    for (int r = 0; r < 8; ++r) {
      float v = acc[nb][r];
      tT[wave][(r + 8 * hf) * H + nb * 16 + l16] = f2bf(v > 0.0f ? v : 0.0f);
    }
  __syncthreads();

  // layer 2: out = relu(t @ w2 + b2)
  #pragma unroll
  for (int nb = 0; nb < 4; ++nb) {
    float bias = b2[nb * 16 + l16];
    #pragma unroll
    for (int r = 0; r < 8; ++r) acc[nb][r] = bias;
    #pragma unroll
    for (int kb = 0; kb < 2; ++kb) {
      FragBF A, B;
      #pragma unroll
      for (int e2 = 0; e2 < 16; ++e2) {
        int k = kb * 32 + kmap(e2, hf);
        A.u[e2] = tT[wave][l16 * H + k];
        B.u[e2] = w2t[(nb * 16 + l16) * H + k];
      }
      acc[nb] = wmma_bf16(A, B, acc[nb]);
    }
  }
  #pragma unroll
  for (int nb = 0; nb < 4; ++nb) {
    int n = nb * 16 + l16;
    #pragma unroll
    for (int r = 0; r < 8; ++r) {
      long long node = n0 + r + 8 * hf;
      if (node < N) {
        float v = acc[nb][r];
        hout[node * H + n] = v > 0.0f ? v : 0.0f;
      }
    }
  }
}

// ---------------------------------------------------------------------------
// classifier: log_softmax(h @ lin_w + lin_b). One 64-thread block per node.
// ---------------------------------------------------------------------------
__global__ void __launch_bounds__(64)
classifier(const float* __restrict__ h, const float* __restrict__ w,
           const float* __restrict__ b, float* __restrict__ out, int C) {
  __shared__ float hrow[H];
  __shared__ float logits[48];
  __shared__ float red[2];
  const long long node = blockIdx.x;
  const int t = threadIdx.x;
  hrow[t] = h[node * H + t];
  __syncthreads();
  if (t < C) {
    float s = b[t];
    #pragma unroll 8
    for (int k = 0; k < H; ++k) s = fmaf(hrow[k], w[k * C + t], s);
    logits[t] = s;
  }
  __syncthreads();
  if (t == 0) {
    float mx = logits[0];
    for (int i = 1; i < C; ++i) mx = fmaxf(mx, logits[i]);
    float se = 0.0f;
    for (int i = 0; i < C; ++i) se += expf(logits[i] - mx);
    red[0] = mx; red[1] = logf(se);
  }
  __syncthreads();
  if (t < C) out[node * C + t] = logits[t] - red[0] - red[1];
}

// ---------------------------------------------------------------------------
extern "C" void kernel_launch(void* const* d_in, const int* in_sizes, int n_in,
                              void* d_out, int out_size, void* d_ws, size_t ws_size,
                              hipStream_t stream) {
  const float* x     = (const float*)d_in[0];
  const int*   eidx  = (const int*)  d_in[1];
  const float* eattr = (const float*)d_in[2];
  const float* e_w1  = (const float*)d_in[3];
  const float* e_b1  = (const float*)d_in[4];
  const float* e_w2  = (const float*)d_in[5];
  const float* e_b2  = (const float*)d_in[6];
  const float* eps1  = (const float*)d_in[7];
  const float* m1_w1 = (const float*)d_in[8];
  const float* m1_b1 = (const float*)d_in[9];
  const float* m1_w2 = (const float*)d_in[10];
  const float* m1_b2 = (const float*)d_in[11];
  const float* eps2  = (const float*)d_in[12];
  const float* m2_w1 = (const float*)d_in[13];
  const float* m2_b1 = (const float*)d_in[14];
  const float* m2_w2 = (const float*)d_in[15];
  const float* m2_b2 = (const float*)d_in[16];
  const float* lin_w = (const float*)d_in[17];
  const float* lin_b = (const float*)d_in[18];

  const int E = in_sizes[2];              // edge_attr is [E,1]
  const int N = in_sizes[0] / H;          // x is [N,64]
  const int C = in_sizes[18];             // lin_b is [C]
  const int* src = eidx;
  const int* dst = eidx + E;

  float* aggr = (float*)d_ws;
  float* h1   = aggr + (size_t)N * H;
  float* h2   = h1   + (size_t)N * H;

  const long long nh = (long long)N * H;
  const int zgrid = (int)((nh + 255) / 256);
  const int egrid = (E + BLOCK_M - 1) / BLOCK_M;
  const int ngrid = (N + BLOCK_M - 1) / BLOCK_M;

  // ---- conv1 ----
  zero_f32<<<zgrid, 256, 0, stream>>>(aggr, nh);
  edge_msg_scatter<<<egrid, 256, 0, stream>>>(x, src, dst, eattr,
                                              e_w1, e_b1, e_w2, e_b2, aggr, E);
  node_mlp<<<ngrid, 256, 0, stream>>>(x, aggr, eps1,
                                      m1_w1, m1_b1, m1_w2, m1_b2, h1, N);
  // ---- conv2 ----
  zero_f32<<<zgrid, 256, 0, stream>>>(aggr, nh);
  edge_msg_scatter<<<egrid, 256, 0, stream>>>(h1, src, dst, eattr,
                                              e_w1, e_b1, e_w2, e_b2, aggr, E);
  node_mlp<<<ngrid, 256, 0, stream>>>(h1, aggr, eps2,
                                      m2_w1, m2_b1, m2_w2, m2_b2, h2, N);
  // ---- classifier + log_softmax ----
  classifier<<<N, 64, 0, stream>>>(h2, lin_w, lin_b, (float*)d_out, C);
}
